// RegionSelfAttention_86492051406968
// MI455X (gfx1250) — matrix operand order repaired and checked by reading, not compiled, not run on hardware
//
#include <hip/hip_runtime.h>

// Problem constants (match reference)
#define B_SZ 32
#define R_SZ 4096
#define D_SZ 512
#define H_SZ 512
#define RSPLIT 8            // R-chunks per batch for the main pass (512 rows each)
#define RT 64               // rows (r) per iteration in main pass
#define LDS_STRIDE_U32 260  // (512 bf16 elems = 256 dwords) + 4 dword pad

typedef __attribute__((ext_vector_type(16))) __bf16 v16bf;
typedef __attribute__((ext_vector_type(8)))  float  v8f;

union Frag16 { v16bf v; unsigned u[8]; };

// float -> bf16 bits, round-to-nearest-even
__device__ __forceinline__ unsigned short f2bf(float f) {
  unsigned u = __float_as_uint(f);
  u += 0x7fffu + ((u >> 16) & 1u);
  return (unsigned short)(u >> 16);
}
__device__ __forceinline__ unsigned pack_bf16(float a, float b) {
  return (unsigned)f2bf(a) | ((unsigned)f2bf(b) << 16);
}
// A-frag (16x32 bf16) K-pair base for VGPR j (ISA 7.12.2): lanes0-15 j<4 -> 2j,
// j>=4 -> 16+2(j-4); lanes16-31 add +8.
__device__ __forceinline__ int kpA(int j, int half) {
  return 2 * j + (j >= 4 ? 8 : 0) + 8 * half;
}
// async copy 4 bytes global -> LDS (CDNA5 async path, tracked by ASYNCcnt)
__device__ __forceinline__ void async_b32_to_lds(void* lds_ptr, const void* gptr) {
  unsigned lds_off = (unsigned)(unsigned long long)lds_ptr;  // low 32 bits = LDS byte addr
  asm volatile("global_load_async_to_lds_b32 %0, %1, off"
               :: "v"(lds_off), "v"(gptr) : "memory");
}
__device__ __forceinline__ void wait_asynccnt0() {
#if __has_builtin(__builtin_amdgcn_s_wait_asynccnt)
  __builtin_amdgcn_s_wait_asynccnt(0);
#else
  asm volatile("s_wait_asynccnt 0x0" ::: "memory");
#endif
}

// ---------------- K0: Wv f32 -> bf16 (packed pairs) ----------------
__global__ __launch_bounds__(256) void cvt_wv_kernel(const float* __restrict__ src,
                                                     unsigned* __restrict__ dst) {
  int i = blockIdx.x * 256 + threadIdx.x;   // pair index, n = 512*512/2
  float2 v = ((const float2*)src)[i];
  dst[i] = pack_bf16(v.x, v.y);
}

// ---------------- K1/K2: C[32][512] = A[32][512] @ (BT ? W^T : W) ----------------
template <bool BT>
__global__ __launch_bounds__(128) void gemm32_kernel(const float* __restrict__ A,
                                                     const float* __restrict__ W,
                                                     float* __restrict__ C) {
  int tid  = threadIdx.x;
  int wave = tid >> 5, lane = tid & 31;
  int half = lane >> 4, l16 = lane & 15;
  int t  = blockIdx.x * 4 + wave;       // 64 tiles: 2 (M) x 32 (N)
  int m0 = (t >> 5) * 16;
  int n0 = (t & 31) * 16;
  v8f acc = {0.f, 0.f, 0.f, 0.f, 0.f, 0.f, 0.f, 0.f};
#pragma unroll 1
  for (int ks = 0; ks < 16; ks++) {
    int k0 = ks * 32;
    Frag16 a, b;
#pragma unroll
    for (int j = 0; j < 8; j++) {
      int ka = k0 + 2 * j + 16 * half;
      float2 av = *(const float2*)(A + (m0 + l16) * D_SZ + ka);
      a.u[j] = pack_bf16(av.x, av.y);
      int kb = k0 + 2 * j + 16 * half;  // B-frag: VGPR j holds K=2j,2j+1 (+16 for hi lanes)
      float b0, b1;
      if (BT) { b0 = W[(n0 + l16) * D_SZ + kb];     b1 = W[(n0 + l16) * D_SZ + kb + 1]; }
      else    { b0 = W[kb * H_SZ + n0 + l16];       b1 = W[(kb + 1) * H_SZ + n0 + l16]; }
      b.u[j] = pack_bf16(b0, b1);
    }
    acc = __builtin_amdgcn_wmma_f32_16x16x32_bf16(false, a.v, false, b.v,
                                                  (short)0, acc, false, false);
  }
#pragma unroll
  for (int v = 0; v < 8; v++) {
    int m = m0 + 8 * half + v;          // C/D layout: VGPR v -> M = v + 8*half
    C[m * H_SZ + n0 + l16] = acc[v];
  }
}

// ---------------- K3: dist[b][r] = rf[b,r,:] . qk[b,:] ----------------
__global__ __launch_bounds__(256) void dist_kernel(const float* __restrict__ rf,
                                                   const float* __restrict__ qk,
                                                   float* __restrict__ dist) {
  __shared__ float sq[D_SZ];
  int b    = blockIdx.x / (R_SZ / 8);
  int rblk = blockIdx.x % (R_SZ / 8);
  int tid  = threadIdx.x;
  for (int i = tid; i < D_SZ; i += 256) sq[i] = qk[b * D_SZ + i];
  __syncthreads();
  int wave = tid >> 5, lane = tid & 31;
  int r = rblk * 8 + wave;
  const float4* row = (const float4*)(rf + ((size_t)b * R_SZ + r) * D_SZ);
  const float4* q4  = (const float4*)sq;
  float acc = 0.f;
#pragma unroll
  for (int i = 0; i < 4; i++) {
    float4 x = row[lane + 32 * i];
    float4 y = q4[lane + 32 * i];
    acc += x.x * y.x + x.y * y.y + x.z * y.z + x.w * y.w;
  }
#pragma unroll
  for (int off = 16; off; off >>= 1) acc += __shfl_xor(acc, off, 32);
  if (lane == 0) dist[b * R_SZ + r] = acc;
}

// ---------------- K4: masked softmax in place over R ----------------
__global__ __launch_bounds__(256) void softmax_kernel(float* __restrict__ score,
                                                      const int* __restrict__ sq_len) {
  int b = blockIdx.x;
  int L = sq_len[b];
  float* x = score + b * R_SZ;
  int tid = threadIdx.x;
  __shared__ float red[256];
  float xs[16];
  float m = -3.4e38f;
#pragma unroll
  for (int i = 0; i < 16; i++) {
    int r = tid + 256 * i;
    float v = x[r];
    xs[i] = v;
    if (r < L && v > m) m = v;
  }
  red[tid] = m; __syncthreads();
  for (int s = 128; s; s >>= 1) { if (tid < s) red[tid] = fmaxf(red[tid], red[tid + s]); __syncthreads(); }
  m = red[0]; __syncthreads();
  const float scale = 0.044194173824159216f;  // 1/sqrt(512)
  float es[16];
  float ssum = 0.f;
#pragma unroll
  for (int i = 0; i < 16; i++) {
    int r = tid + 256 * i;
    float e = (r < L) ? __expf((xs[i] - m) * scale) : 0.f;
    es[i] = e; ssum += e;
  }
  red[tid] = ssum; __syncthreads();
  for (int s = 128; s; s >>= 1) { if (tid < s) red[tid] += red[tid + s]; __syncthreads(); }
  float inv = 1.f / red[0];
#pragma unroll
  for (int i = 0; i < 16; i++) { int r = tid + 256 * i; x[r] = es[i] * inv; }
}

// ---------------- K5: out_partial[s][b][h] = sum_r score * relu(rf @ Wv^T) ----------------
// 512 threads = 16 waves; each wave owns 2 n-tiles (32 cols); 64 rf rows staged per
// iteration (4 row-tiles) so each Wv B-fragment feeds 4 WMMAs.
__global__ __launch_bounds__(512) void kv_attn_kernel(const float* __restrict__ rf,
                                                      const unsigned short* __restrict__ wv_bf,
                                                      const float* __restrict__ score,
                                                      float* __restrict__ partial) {
  __shared__ unsigned sA[RT * LDS_STRIDE_U32];  // 64 rows x 512 bf16 (padded) = 66.6 KB
  __shared__ float sScore[RT];
  int b = blockIdx.x;
  int s = blockIdx.y;
  int tid  = threadIdx.x;
  int wave = tid >> 5, lane = tid & 31;
  int half = lane >> 4, l16 = lane & 15;
  const float* rfb = rf + (size_t)b * R_SZ * D_SZ;
  const float* scb = score + b * R_SZ;
  int n_base = wave * 32;                 // each wave owns 2 n-tiles of 16 cols
  float accN[2] = {0.f, 0.f};
  int r_beg = s * (R_SZ / RSPLIT);
  int r_end = r_beg + (R_SZ / RSPLIT);
  int stg_row = tid >> 3;                 // 64 rows, 8 threads/row
  int stg_col = (tid & 7) * 64;           // 64 floats per thread
  // per-lane hoisted bases
  const unsigned* aRow0 = &sA[l16 * LDS_STRIDE_U32];
  const unsigned short* wvb0 = wv_bf + (size_t)(n_base + l16) * D_SZ + 16 * half;
  const unsigned short* wvb1 = wv_bf + (size_t)(n_base + 16 + l16) * D_SZ + 16 * half;
#pragma unroll 1
  for (int r0 = r_beg; r0 < r_end; r0 += RT) {
    // --- score chunk via CDNA5 async global->LDS copy (overlaps with staging) ---
    if (tid < RT) async_b32_to_lds(&sScore[tid], scb + r0 + tid);
    // --- stage rf tile (RT x 512) as bf16 into LDS ---
    {
      const float4* src = (const float4*)(rfb + (size_t)(r0 + stg_row) * D_SZ + stg_col);
      unsigned* dst = &sA[stg_row * LDS_STRIDE_U32 + (stg_col >> 1)];
#pragma unroll
      for (int i = 0; i < 16; i++) {
        float4 v = src[i];
        dst[2 * i]     = pack_bf16(v.x, v.y);
        dst[2 * i + 1] = pack_bf16(v.z, v.w);
      }
      if (r0 + RT < r_end)  // prefetch next tile
        __builtin_prefetch(rfb + (size_t)(r0 + RT + stg_row) * D_SZ + stg_col, 0, 1);
    }
    wait_asynccnt0();
    __syncthreads();
    v8f acc[2][4];
#pragma unroll
    for (int t = 0; t < 2; t++)
#pragma unroll
      for (int rt = 0; rt < 4; rt++)
        acc[t][rt] = (v8f){0.f,0.f,0.f,0.f,0.f,0.f,0.f,0.f};
#pragma unroll 1
    for (int ks = 0; ks < 16; ks++) {
      int k0 = ks * 32;
      // hoist both B-fragments for this k-step (reused by 4 row-tiles)
      Frag16 bf0, bf1;
      {
        const uint4* bp = (const uint4*)(wvb0 + k0);
        uint4 q0 = bp[0], q1 = bp[1];
        bf0.u[0] = q0.x; bf0.u[1] = q0.y; bf0.u[2] = q0.z; bf0.u[3] = q0.w;
        bf0.u[4] = q1.x; bf0.u[5] = q1.y; bf0.u[6] = q1.z; bf0.u[7] = q1.w;
        bp = (const uint4*)(wvb1 + k0);
        q0 = bp[0]; q1 = bp[1];
        bf1.u[0] = q0.x; bf1.u[1] = q0.y; bf1.u[2] = q0.z; bf1.u[3] = q0.w;
        bf1.u[4] = q1.x; bf1.u[5] = q1.y; bf1.u[6] = q1.z; bf1.u[7] = q1.w;
      }
#pragma unroll
      for (int rt = 0; rt < 4; rt++) {
        Frag16 a;
#pragma unroll
        for (int j = 0; j < 8; j++) {
          int kofs = (k0 + 2 * j + 16 * half) >> 1;
          a.u[j] = aRow0[rt * 16 * LDS_STRIDE_U32 + kofs];
        }
        acc[0][rt] = __builtin_amdgcn_wmma_f32_16x16x32_bf16(false, a.v, false, bf0.v,
                                                             (short)0, acc[0][rt], false, false);
        acc[1][rt] = __builtin_amdgcn_wmma_f32_16x16x32_bf16(false, a.v, false, bf1.v,
                                                             (short)0, acc[1][rt], false, false);
      }
    }
    // --- relu + score weighting ---
#pragma unroll
    for (int t = 0; t < 2; t++)
#pragma unroll
      for (int rt = 0; rt < 4; rt++)
#pragma unroll
        for (int v = 0; v < 8; v++) {
          int m = rt * 16 + 8 * half + v;
          accN[t] += sScore[m] * fmaxf(acc[t][rt][v], 0.f);
        }
    __syncthreads();
  }
#pragma unroll
  for (int t = 0; t < 2; t++) {
    float tot = accN[t] + __shfl_xor(accN[t], 16, 32);
    if (half == 0)
      partial[((size_t)s * B_SZ + b) * H_SZ + n_base + t * 16 + l16] = tot;
  }
}

// ---------------- K6: out = query + sum_s partial[s] ----------------
__global__ __launch_bounds__(256) void finalize_kernel(const float* __restrict__ query,
                                                       const float* __restrict__ partial,
                                                       float* __restrict__ out) {
  int i = blockIdx.x * 256 + threadIdx.x;  // 0 .. B*H-1
  float v = query[i];
#pragma unroll
  for (int s = 0; s < RSPLIT; s++) v += partial[s * (B_SZ * H_SZ) + i];
  out[i] = v;
}

extern "C" void kernel_launch(void* const* d_in, const int* in_sizes, int n_in,
                              void* d_out, int out_size, void* d_ws, size_t ws_size,
                              hipStream_t stream) {
  const float* seq = (const float*)d_in[0];   // [32,512]
  const float* rf  = (const float*)d_in[1];   // [32,4096,512]
  const float* Wq  = (const float*)d_in[2];   // [512,512]
  const float* Wk  = (const float*)d_in[3];   // [512,512]
  const float* Wv  = (const float*)d_in[4];   // [512,512]
  const int*   sqL = (const int*)d_in[5];     // [32]

  float* out   = (float*)d_out;               // [32,512]
  float* score = out + B_SZ * H_SZ;           // [32,4096]

  // workspace layout
  float* q_ws  = (float*)d_ws;                            // 16384 f32
  float* qk_ws = q_ws + B_SZ * H_SZ;                      // 16384 f32
  unsigned short* wv_bf = (unsigned short*)(qk_ws + B_SZ * D_SZ);  // 512*512 bf16
  float* partial = (float*)((char*)wv_bf + (size_t)H_SZ * D_SZ * 2); // RSPLIT*16384 f32

  // K0: Wv -> bf16
  cvt_wv_kernel<<<(H_SZ * D_SZ / 2) / 256, 256, 0, stream>>>(Wv, (unsigned*)wv_bf);
  // K1: query = seq @ Wq^T
  gemm32_kernel<true><<<16, 128, 0, stream>>>(seq, Wq, q_ws);
  // K2: qk = query @ Wk   (collapses the keyp GEMM)
  gemm32_kernel<false><<<16, 128, 0, stream>>>(q_ws, Wk, qk_ws);
  // K3: raw dist into score region
  dist_kernel<<<B_SZ * (R_SZ / 8), 256, 0, stream>>>(rf, qk_ws, score);
  // K4: masked softmax in place
  softmax_kernel<<<B_SZ, 256, 0, stream>>>(score, sqL);
  // K5: fused value GEMM + weighted reduction
  dim3 g5(B_SZ, RSPLIT);
  kv_attn_kernel<<<g5, 512, 0, stream>>>(rf, wv_bf, score, partial);
  // K6: add query and partials
  finalize_kernel<<<(B_SZ * H_SZ) / 256, 256, 0, stream>>>(q_ws, partial, out);
}
